// Diffusion_89077621719508
// MI455X (gfx1250) — compile-verified, matrix operand order checked by reading, not run on hardware
//
#include <hip/hip_runtime.h>
#include <hip/hip_bf16.h>

#define BB 32
#define NN 128
#define SS 64
#define DD 512

typedef __attribute__((ext_vector_type(8)))  __bf16 v8bf;
typedef __attribute__((ext_vector_type(16))) __bf16 v16bf;
typedef __attribute__((ext_vector_type(8)))  float  v8f;

__device__ inline float wave_reduce_sum(float v) {
  #pragma unroll
  for (int off = 16; off > 0; off >>= 1) v += __shfl_xor(v, off, 32);
  return v;
}

// ---------------- weight convert: Wd1[:, 512:1024] -> bf16 row-major [1024][512]
__global__ void k_wd1_bf16(const float* __restrict__ Wd1, __bf16* __restrict__ wd1b) {
  int idx = blockIdx.x * blockDim.x + threadIdx.x;   // 1024*512
  int j = idx >> 9, d = idx & 511;
  wd1b[idx] = (__bf16)Wd1[(size_t)j * 1024 + 512 + d];
}

// ---------------- cond layer1: h1 = silu(pe[t] @ Wt1^T + bt1)
__global__ void k_cond1(const int* __restrict__ ts, const float* __restrict__ pe,
                        const float* __restrict__ Wt1, const float* __restrict__ bt1,
                        float* __restrict__ h1) {
  int b = blockIdx.x;
  int j = blockIdx.y * blockDim.x + threadIdx.x;
  const float* pr = pe + (size_t)ts[b] * DD;
  const float* wr = Wt1 + (size_t)j * DD;
  float acc = 0.f;
  for (int d = 0; d < DD; d += 4) {
    float4 a = *(const float4*)&pr[d];
    float4 w = *(const float4*)&wr[d];
    acc += a.x * w.x + a.y * w.y + a.z * w.z + a.w * w.w;
  }
  acc += bt1[j];
  h1[(size_t)b * DD + j] = acc / (1.f + expf(-acc));
}

// ---------------- generic y[b,e] = sum_d in[b,d]*W[e*ldw+d] (+bias)
__global__ void k_vecmat(const float* __restrict__ in, const float* __restrict__ W, int ldw,
                         const float* __restrict__ bias, float* __restrict__ out, int odim) {
  int b = blockIdx.x;
  int e = blockIdx.y * blockDim.x + threadIdx.x;
  const float* ir = in + (size_t)b * DD;
  const float* wr = W + (size_t)e * ldw;
  float acc = 0.f;
  for (int d = 0; d < DD; d += 4) {
    float4 a = *(const float4*)&ir[d];
    float4 w = *(const float4*)&wr[d];
    acc += a.x * w.x + a.y * w.y + a.z * w.z + a.w * w.w;
  }
  if (bias) acc += bias[e];
  out[(size_t)b * odim + e] = acc;
}

// ---------------- qk[b,e] = sum_d q[b,d]*Wk[d*512+e]   (W column access, coalesced across e)
__global__ void k_vecmat_col(const float* __restrict__ q, const float* __restrict__ Wk,
                             float* __restrict__ qk) {
  int b = blockIdx.x;
  int e = blockIdx.y * blockDim.x + threadIdx.x;
  const float* qr = q + (size_t)b * DD;
  float acc = 0.f;
  for (int d = 0; d < DD; ++d) acc += qr[d] * Wk[(size_t)d * DD + e];
  qk[(size_t)b * DD + e] = acc;
}

// ---------------- qin = fusion + cond
__global__ void k_add(const float* __restrict__ a, const float* __restrict__ b,
                      float* __restrict__ o) {
  int i = blockIdx.x * blockDim.x + threadIdx.x;
  o[i] = a[i] + b[i];
}

// ---------------- sim -> argmax over S -> gather tif (f32 + bf16). Bandwidth kernel: 536MB stream.
__global__ __launch_bounds__(256) void k_simgather(const float* __restrict__ fusion,
                                                   const float* __restrict__ tar,
                                                   float* __restrict__ tiff,
                                                   __bf16* __restrict__ tifb) {
  const int blk = blockIdx.x;            // b*128 + n
  const int b = blk >> 7;
  const int tid = threadIdx.x, lane = tid & 31, wave = tid >> 5;
  __shared__ float ssim[SS];
  __shared__ int sbest;

  // cache fusion chunk per lane: 16 floats
  float4 f4[4];
  #pragma unroll
  for (int c = 0; c < 4; ++c)
    f4[c] = *(const float4*)&fusion[(size_t)b * DD + lane * 4 + c * 128];

  const float* base = tar + (size_t)blk * SS * DD;
  for (int s = wave; s < SS; s += 8) {
    const float* tr = base + (size_t)s * DD;
    float acc = 0.f;
    #pragma unroll
    for (int c = 0; c < 4; ++c) {
      float4 t4 = *(const float4*)&tr[lane * 4 + c * 128];
      acc += f4[c].x * t4.x + f4[c].y * t4.y + f4[c].z * t4.z + f4[c].w * t4.w;
    }
    acc = wave_reduce_sum(acc);
    if (lane == 0) ssim[s] = acc;
  }
  __syncthreads();
  if (tid == 0) {
    float best = ssim[0]; int bi = 0;
    for (int s = 1; s < SS; ++s) { float v = ssim[s]; if (v > best) { best = v; bi = s; } }
    sbest = bi;
  }
  __syncthreads();
  const float* br = base + (size_t)sbest * DD;
  for (int d = tid; d < DD; d += 256) {
    float v = br[d];
    tiff[(size_t)blk * DD + d] = v;
    tifb[(size_t)blk * DD + d] = (__bf16)v;
  }
}

// ---------------- folded attention: logits->softmax->w, ubar = sum_n w*(tif+cond)
__global__ __launch_bounds__(256) void k_attn(const float* __restrict__ qk,
                                              const float* __restrict__ tiff,
                                              const float* __restrict__ cond,
                                              const float* __restrict__ x,
                                              float* __restrict__ wbuf,
                                              float* __restrict__ ubar) {
  const int b = blockIdx.x;
  const int tid = threadIdx.x, lane = tid & 31, wave = tid >> 5;
  __shared__ float slog[NN];
  __shared__ float red[NN];
  __shared__ float sw[NN];
  const float* qkb = qk + (size_t)b * DD;
  const float* cb = cond + (size_t)b * DD;

  for (int nn = 0; nn < 16; ++nn) {
    int n = wave * 16 + nn;
    const float* tr = tiff + ((size_t)b * NN + n) * DD;
    float acc = 0.f;
    #pragma unroll
    for (int c = 0; c < 4; ++c) {
      int d = lane * 4 + c * 128;
      float4 qv = *(const float4*)&qkb[d];
      float4 cv = *(const float4*)&cb[d];
      float4 tv = *(const float4*)&tr[d];
      acc += qv.x * (tv.x + cv.x) + qv.y * (tv.y + cv.y) +
             qv.z * (tv.z + cv.z) + qv.w * (tv.w + cv.w);
    }
    acc = wave_reduce_sum(acc);
    if (lane == 0) slog[n] = acc + x[(size_t)b * NN + n];
  }
  __syncthreads();
  if (tid < NN) red[tid] = slog[tid];
  __syncthreads();
  for (int s = 64; s > 0; s >>= 1) { if (tid < s) red[tid] = fmaxf(red[tid], red[tid + s]); __syncthreads(); }
  float mx = red[0];
  __syncthreads();
  if (tid < NN) { float e = expf(slog[tid] - mx); sw[tid] = e; red[tid] = e; }
  __syncthreads();
  for (int s = 64; s > 0; s >>= 1) { if (tid < s) red[tid] += red[tid + s]; __syncthreads(); }
  float inv = 1.f / red[0];
  __syncthreads();
  if (tid < NN) { float wv = sw[tid] * inv; sw[tid] = wv; wbuf[(size_t)b * NN + tid] = wv; }
  __syncthreads();
  for (int d = tid; d < DD; d += 256) {
    float cd = cb[d];
    const float* tc = tiff + (size_t)b * NN * DD + d;
    float acc = 0.f;
    for (int n = 0; n < NN; ++n) acc += sw[n] * (tc[(size_t)n * DD] + cd);
    ubar[(size_t)b * DD + d] = acc;
  }
}

// ---------------- WMMA helpers: 16-bit A/B fragment loader per ISA VGPR layout
__device__ inline v16bf load_frag_bf16(const __bf16* p) {
  v8bf lo = *(const v8bf*)(p);
  v8bf hi = *(const v8bf*)(p + 16);
  v16bf r;
  #pragma unroll
  for (int i = 0; i < 8; ++i) { r[i] = lo[i]; r[i + 8] = hi[i]; }
  return r;
}

// ---------------- tifpart[4096,1024] = tif_bf16[4096,512] @ wd1b[1024,512]^T (f32 accum)
// Software-pipelined: iteration k+1's fragments are loaded before iteration k's WMMAs
// consume, so the wave has a full K-step of v_wmma work between load issue and wait.
__global__ __launch_bounds__(256) void k_tifpart_wmma(const __bf16* __restrict__ tifb,
                                                      const __bf16* __restrict__ wd1b,
                                                      float* __restrict__ outp) {
  const int tid = threadIdx.x;
  const int lane = tid & 31;
  const int wave = tid >> 5;
  const int wm = wave >> 2;                  // 0..1
  const int wn = wave & 3;                   // 0..3
  const int row0 = blockIdx.x * 32 + wm * 16;
  const int col0 = blockIdx.y * 256 + wn * 64;
  const int m = lane & 15;
  const int hi = lane >> 4;

  v8f acc0 = {}, acc1 = {}, acc2 = {}, acc3 = {};
  const __bf16* arow = tifb + (size_t)(row0 + m) * DD + hi * 8;
  const __bf16* b0 = wd1b + (size_t)(col0 +  0 + m) * DD + hi * 8;
  const __bf16* b1 = wd1b + (size_t)(col0 + 16 + m) * DD + hi * 8;
  const __bf16* b2 = wd1b + (size_t)(col0 + 32 + m) * DD + hi * 8;
  const __bf16* b3 = wd1b + (size_t)(col0 + 48 + m) * DD + hi * 8;

  // prologue
  v16bf ac  = load_frag_bf16(arow);
  v16bf b0c = load_frag_bf16(b0);
  v16bf b1c = load_frag_bf16(b1);
  v16bf b2c = load_frag_bf16(b2);
  v16bf b3c = load_frag_bf16(b3);

  #pragma unroll
  for (int k0 = 0; k0 < DD; k0 += 32) {
    const int k1 = k0 + 32;
    v16bf an, b0n, b1n, b2n, b3n;
    if (k1 < DD) {                    // prefetch next K-step (resolved at compile time)
      an  = load_frag_bf16(arow + k1);
      b0n = load_frag_bf16(b0 + k1);
      b1n = load_frag_bf16(b1 + k1);
      b2n = load_frag_bf16(b2 + k1);
      b3n = load_frag_bf16(b3 + k1);
    } else {
      an = ac; b0n = b0c; b1n = b1c; b2n = b2c; b3n = b3c;
    }
    acc0 = __builtin_amdgcn_wmma_f32_16x16x32_bf16(false, ac, false, b0c, (short)0, acc0, false, false);
    acc1 = __builtin_amdgcn_wmma_f32_16x16x32_bf16(false, ac, false, b1c, (short)0, acc1, false, false);
    acc2 = __builtin_amdgcn_wmma_f32_16x16x32_bf16(false, ac, false, b2c, (short)0, acc2, false, false);
    acc3 = __builtin_amdgcn_wmma_f32_16x16x32_bf16(false, ac, false, b3c, (short)0, acc3, false, false);
    ac = an; b0c = b0n; b1c = b1n; b2c = b2n; b3c = b3n;
  }

  #pragma unroll
  for (int r = 0; r < 8; ++r) {
    size_t row = (size_t)(row0 + r + hi * 8) * 1024;
    outp[row + col0 +  0 + m] = acc0[r];
    outp[row + col0 + 16 + m] = acc1[r];
    outp[row + col0 + 32 + m] = acc2[r];
    outp[row + col0 + 48 + m] = acc3[r];
  }
}

// ---------------- p = wd2 . relu(A[b] + tifpart[b,n]) + bd2 + w ; sort desc over N=128
__global__ __launch_bounds__(256) void k_final(const float* __restrict__ Amat,
                                               const float* __restrict__ tifpart,
                                               const float* __restrict__ Wd2,
                                               const float* __restrict__ bd2,
                                               const float* __restrict__ wbuf,
                                               float* __restrict__ out) {
  const int b = blockIdx.x;
  const int tid = threadIdx.x, lane = tid & 31, wave = tid >> 5;
  __shared__ __align__(16) float sA[1024];
  __shared__ __align__(16) float sW2[1024];
  __shared__ float sp[NN];
  for (int i = tid; i < 1024; i += 256) { sA[i] = Amat[(size_t)b * 1024 + i]; sW2[i] = Wd2[i]; }
  __syncthreads();
  float bd2v = bd2[0];
  for (int nn = 0; nn < 16; ++nn) {
    int n = wave * 16 + nn;
    const float* tp = tifpart + ((size_t)b * NN + n) * 1024;
    float acc = 0.f;
    #pragma unroll
    for (int c = 0; c < 8; ++c) {
      int j = lane * 4 + c * 128;
      float4 a4 = *(const float4*)&sA[j];
      float4 w4 = *(const float4*)&sW2[j];
      float4 t4 = *(const float4*)&tp[j];
      acc += w4.x * fmaxf(a4.x + t4.x, 0.f) + w4.y * fmaxf(a4.y + t4.y, 0.f) +
             w4.z * fmaxf(a4.z + t4.z, 0.f) + w4.w * fmaxf(a4.w + t4.w, 0.f);
    }
    acc = wave_reduce_sum(acc);
    if (lane == 0) sp[n] = -(acc + bd2v + wbuf[(size_t)b * NN + n]);  // negate -> ascending sort
  }
  __syncthreads();
  // bitonic ascending on 128 negated values
  for (int k = 2; k <= NN; k <<= 1) {
    for (int j = k >> 1; j > 0; j >>= 1) {
      if (tid < NN) {
        int ixj = tid ^ j;
        if (ixj > tid) {
          bool up = ((tid & k) == 0);
          float a = sp[tid], c = sp[ixj];
          bool sw = up ? (a > c) : (a < c);
          if (sw) { sp[tid] = c; sp[ixj] = a; }
        }
      }
      __syncthreads();
    }
  }
  if (tid < NN) out[(size_t)b * NN + tid] = -sp[tid];
}

extern "C" void kernel_launch(void* const* d_in, const int* in_sizes, int n_in,
                              void* d_out, int out_size, void* d_ws, size_t ws_size,
                              hipStream_t stream) {
  (void)in_sizes; (void)n_in; (void)out_size; (void)ws_size;
  const float* x      = (const float*)d_in[0];
  const int*   ts     = (const int*)d_in[1];
  const float* fusion = (const float*)d_in[2];
  const float* tar    = (const float*)d_in[3];
  const float* pe     = (const float*)d_in[4];
  const float* Wq     = (const float*)d_in[5];
  const float* Wk     = (const float*)d_in[6];
  const float* Wv     = (const float*)d_in[7];
  const float* Wp     = (const float*)d_in[8];
  const float* bp     = (const float*)d_in[9];
  const float* Wt1    = (const float*)d_in[10];
  const float* bt1    = (const float*)d_in[11];
  const float* Wt2    = (const float*)d_in[12];
  const float* bt2    = (const float*)d_in[13];
  const float* Wd1    = (const float*)d_in[14];
  const float* bd1    = (const float*)d_in[15];
  const float* Wd2    = (const float*)d_in[16];
  const float* bd2    = (const float*)d_in[17];
  float* out = (float*)d_out;

  float* wsf     = (float*)d_ws;
  float* cond    = wsf;                     // 16384
  float* h1      = cond + 16384;            // 16384 (also reused as qin)
  float* qv      = h1 + 16384;              // 16384
  float* qkv     = qv + 16384;              // 16384
  float* tmp_t   = qkv + 16384;             // 16384
  float* new_emb = tmp_t + 16384;           // 16384
  float* ubar    = new_emb + 16384;         // 16384
  float* Amat    = ubar + 16384;            // 32768
  float* wbuf    = Amat + 32768;            // 4096
  float* tiff    = wbuf + 4096;             // 2097152
  float* tifpart = tiff + 2097152;          // 4194304
  __bf16* tifb   = (__bf16*)(tifpart + 4194304);  // 2097152 bf16
  __bf16* wd1b   = tifb + 2097152;                // 524288 bf16

  // weight convert + cond MLP
  k_wd1_bf16<<<2048, 256, 0, stream>>>(Wd1, wd1b);
  k_cond1<<<dim3(BB, 4), 128, 0, stream>>>(ts, pe, Wt1, bt1, h1);
  k_vecmat<<<dim3(BB, 4), 128, 0, stream>>>(h1, Wt2, DD, bt2, cond, DD);

  // sim/argmax/gather (bandwidth-dominant stage)
  k_simgather<<<BB * NN, 256, 0, stream>>>(fusion, tar, tiff, tifb);

  // folded attention
  k_add<<<64, 256, 0, stream>>>(fusion, cond, h1);                       // qin = fusion+cond
  k_vecmat<<<dim3(BB, 4), 128, 0, stream>>>(h1, Wq, DD, nullptr, qv, DD);
  k_vecmat_col<<<dim3(BB, 4), 128, 0, stream>>>(qv, Wk, qkv);
  k_attn<<<BB, 256, 0, stream>>>(qkv, tiff, cond, x, wbuf, ubar);

  // new_emb = (ubar @ Wv^T) @ Wp^T + bp ; A = new_emb @ Wd1[:, :512]^T + bd1
  k_vecmat<<<dim3(BB, 4), 128, 0, stream>>>(ubar, Wv, DD, nullptr, tmp_t, DD);
  k_vecmat<<<dim3(BB, 4), 128, 0, stream>>>(tmp_t, Wp, DD, bp, new_emb, DD);
  k_vecmat<<<dim3(BB, 8), 128, 0, stream>>>(new_emb, Wd1, 1024, bd1, Amat, 1024);

  // main WMMA GEMM: tifpart = tif_bf16 @ wd1b^T (software-pipelined)
  k_tifpart_wmma<<<dim3(128, 4), 256, 0, stream>>>(tifb, wd1b, tifpart);

  // final score + descending sort
  k_final<<<BB, 256, 0, stream>>>(Amat, tifpart, Wd2, bd2, wbuf, out);
}
